// EvolvableSNN_38809324487500
// MI455X (gfx1250) — matrix operand order repaired and checked by reading, not compiled, not run on hardware
//
#include <hip/hip_runtime.h>

// ---------------------------------------------------------------------------
// SNN scan for MI455X (gfx1250, wave32, WMMA bf16, TDM async tensor loads)
//
// Per step t:  cur = A_t @ W,  A_t = [input_t[:, :512] | spikes_{t-1}[:,512:]]
//   (columns n<512 only accumulate over k<512)
// W (bf16, transposed to Wt[n][k]) = 32MB -> resident in 192MB L2.
// Persistent kernel, 32 WGs x 256 thr; grid barriers via L2 atomics.
// A panel (16x4096 bf16 = 128KB) broadcast to each WG's LDS via TDM.
// ---------------------------------------------------------------------------

#define T_STEPS 512
#define NB 8
#define NN 4096
#define NIN 512
#define NWG 32           // 32 workgroups * 8 waves * 16 cols = 4096 columns
#define DT_F 0.001f

typedef float  v8f   __attribute__((ext_vector_type(8)));
typedef __bf16 v16bf __attribute__((ext_vector_type(16)));
typedef unsigned int uv4 __attribute__((ext_vector_type(4)));
typedef int          iv4 __attribute__((ext_vector_type(4)));
typedef int          iv8 __attribute__((ext_vector_type(8)));

struct Q16 { unsigned a, b, c, d; };                 // 16B POD load unit
struct Q32 { Q16 lo, hi; };                          // 32B = one bf16 fragment

static __device__ __forceinline__ unsigned short f2bf(float f) {
  unsigned u = __builtin_bit_cast(unsigned, f);
  unsigned r = u + 0x7FFFu + ((u >> 16) & 1u);       // round-to-nearest-even
  return (unsigned short)(r >> 16);
}

static __device__ __forceinline__ v16bf make_frag(Q16 x, Q16 y) {
  Q32 q{x, y};
  return __builtin_bit_cast(v16bf, q);
}

// -------------------------- init kernels -----------------------------------

// input_signal[:, :, :512] (fp32) -> inbf[t][b][k] (bf16), 512*8*512 elems
__global__ void k_init_input(const float* __restrict__ in,
                             unsigned short* __restrict__ inbf) {
  size_t i = (size_t)blockIdx.x * 256 + threadIdx.x;
  if (i < (size_t)T_STEPS * NB * NIN) {
    size_t tb = i >> 9;            // t*8+b
    int k = (int)(i & 511);
    inbf[i] = f2bf(in[tb * NN + k]);
  }
}

// W[k][n] (fp32) -> Wt[n][k] (bf16), tiled transpose
__global__ void k_transpose(const float* __restrict__ W,
                            unsigned short* __restrict__ Wt) {
  __shared__ float tile[32][33];
  int tx = threadIdx.x, ty = threadIdx.y;            // block (32,8)
  int n0 = blockIdx.x * 32, k0 = blockIdx.y * 32;
#pragma unroll
  for (int i = 0; i < 32; i += 8)
    tile[ty + i][tx] = W[(size_t)(k0 + ty + i) * NN + (n0 + tx)];
  __syncthreads();
#pragma unroll
  for (int i = 0; i < 32; i += 8)
    Wt[(size_t)(n0 + ty + i) * NN + (k0 + tx)] = f2bf(tile[tx][ty + i]);
}

// A_global: 16 x 4096 bf16. Rows 0..7: [input_0 | zeros]; rows 8..15: zero.
// Also resets the grid-barrier state.
__global__ void k_init_A(unsigned short* __restrict__ Ag,
                         const unsigned short* __restrict__ inbf,
                         unsigned* __restrict__ bar) {
  int i = blockIdx.x * 256 + threadIdx.x;            // 65536 elems
  if (i < 16 * NN) {
    int b = i >> 12, k = i & (NN - 1);
    unsigned short v = 0;
    if (b < NB && k < NIN) v = inbf[b * NIN + k];    // t = 0
    Ag[i] = v;
  }
  if (i < 2) bar[i] = 0;
}

// -------------------------- grid barrier -----------------------------------

static __device__ __forceinline__ void grid_barrier(unsigned* bar) {
  __syncthreads();
  if (threadIdx.x == 0) {
    __threadfence();
    unsigned gen  = __hip_atomic_load(&bar[1], __ATOMIC_RELAXED, __HIP_MEMORY_SCOPE_AGENT);
    unsigned prev = __hip_atomic_fetch_add(&bar[0], 1u, __ATOMIC_ACQ_REL, __HIP_MEMORY_SCOPE_AGENT);
    if (prev == (unsigned)(NWG - 1)) {
      __hip_atomic_store(&bar[0], 0u, __ATOMIC_RELAXED, __HIP_MEMORY_SCOPE_AGENT);
      __hip_atomic_fetch_add(&bar[1], 1u, __ATOMIC_RELEASE, __HIP_MEMORY_SCOPE_AGENT);
    } else {
      while (__hip_atomic_load(&bar[1], __ATOMIC_ACQUIRE, __HIP_MEMORY_SCOPE_AGENT) == gen)
        __builtin_amdgcn_s_sleep(2);
    }
  }
  __syncthreads();
}

// -------------------------- persistent scan --------------------------------

__global__ void __launch_bounds__(256, 1)
snn_persistent(const unsigned short* __restrict__ inbf,
               const unsigned short* __restrict__ Wt,
               unsigned short* __restrict__ Ag,
               const float* __restrict__ tau_m,
               const float* __restrict__ tau_s,
               const float* __restrict__ thr_p,
               float* __restrict__ out,
               unsigned* __restrict__ bar) {
  const int wg   = blockIdx.x;              // 0..31
  const int wave = threadIdx.x >> 5;        // 0..7
  const int lane = threadIdx.x & 31;
  const int half = lane >> 4;
  const int col  = lane & 15;
  const int n    = wg * 128 + wave * 16 + col;   // output column of this lane
  const bool small = (wg < 4);              // columns n<512: K-loop only 512
  const int kIters = small ? (NIN / 32) : (NN / 32);

  __shared__ __align__(16) unsigned short ldsA[16 * NN];   // 128 KB panel

  const float thr  = thr_p[n];
  const float dsyn = 1.0f - DT_F / tau_s[n];
  const float dmem = 1.0f - DT_F / tau_m[n];

  float syn[8], mem[8];
#pragma unroll
  for (int r = 0; r < 8; ++r) { syn[r] = 0.0f; mem[r] = 0.0f; }

  // ---- TDM descriptor: A_global (16 x 4096 bf16, stride 4096) -> LDS ----
  unsigned lds_off = (unsigned)(unsigned long long)(&ldsA[0]); // low32 = LDS offset
  unsigned long long ga = (unsigned long long)Ag;
  uv4 g0 = { 1u,                                    // count=1 (valid), user mode
             lds_off,                               // lds_addr
             (unsigned)ga,                          // global_addr[31:0]
             ((unsigned)((ga >> 32) & 0x1FFFFFFull)) | (2u << 30) }; // addr[56:32] | type=2
  iv8 g1 = { (int)0x00010000,                       // data_size=1 (2 bytes)
             (int)(4096u << 16),                    // tensor_dim0[15:0]=4096 (hi16 of w1)
             (int)(16u << 16),                      // tensor_dim1[15:0]=16
             (int)(4096u << 16),                    // tile_dim0=4096
             16,                                    // tile_dim1=16, tile_dim2=0
             4096,                                  // tensor_dim0_stride lo32
             0, 0 };
  iv4 g2 = { 0, 0, 0, 0 };
  iv4 g3 = { 0, 0, 0, 0 };

  for (int t = 0; t < T_STEPS; ++t) {
    // snapshot shared activation panel into LDS (one TDM op per WG)
    if (wave == 0) {
#if __clang_major__ >= 23
      iv8 g4 = { 0, 0, 0, 0, 0, 0, 0, 0 };
      __builtin_amdgcn_tensor_load_to_lds(g0, g1, g2, g3, g4, 0);
#else
      __builtin_amdgcn_tensor_load_to_lds(g0, g1, g2, g3, 0);
#endif
      __builtin_amdgcn_s_wait_tensorcnt(0);
    }
    __syncthreads();
    grid_barrier(bar);       // all WGs have their snapshot; A_global writable

    // ---- cur = A(16xK) x Wt(Kx16), K in {512, 4096} ----
    v8f acc = {};
    const Q16* wp = (const Q16*)(Wt + (size_t)n * NN + half * 8);
    const unsigned aBase = (unsigned)col * NN + half * 8;   // A row = col
    for (int kk = 0; kk < kIters; ++kk) {
      const int k0 = kk * 32;
      __builtin_prefetch(wp + 8, 0, 1);                     // global_prefetch
      Q16 a0 = *(const Q16*)&ldsA[aBase + k0];              // K: 8*half+0..7
      Q16 a1 = *(const Q16*)&ldsA[aBase + k0 + 16];         // K: 16+8*half+0..7
      Q16 b0 = wp[0];                                       // K: 8*half+0..7
      Q16 b1 = wp[2];                                       // K: 16+8*half+0..7
      wp += 4;                                              // +32 bf16
      v16bf af = make_frag(a0, a1);
      v16bf bf = make_frag(b0, b1);
      acc = __builtin_amdgcn_wmma_f32_16x16x32_bf16(
          false, af, false, bf, (short)0, acc, false, false);
    }

    // ---- neuron update: rows b = 0..7 (lanes 0-15 hold valid rows) ----
    float spk[8];
#pragma unroll
    for (int r = 0; r < 8; ++r) {
      float s = syn[r] * dsyn + acc[r] * DT_F;
      float m = mem[r] * dmem + s * DT_F;
      float sp = (m >= thr) ? 1.0f : 0.0f;
      syn[r] = s;
      mem[r] = m * (1.0f - sp);
      spk[r] = sp;
    }

    if (half == 0) {
      float* ob = out + (size_t)t * (NB * NN) + n;
#pragma unroll
      for (int r = 0; r < 8; ++r) ob[(size_t)r * NN] = spk[r];
      if (!small) {                       // feedback only for k = n >= 512
#pragma unroll
        for (int r = 0; r < 8; ++r) Ag[r * NN + n] = f2bf(spk[r]);
      }
    }

    // WGs 0..3 (light K-loop) stage input_{t+1} into A_global[:, :512]
    if (small && (t + 1) < T_STEPS) {
      for (int i = threadIdx.x; i < 1024; i += 256) {
        int b  = wg * 2 + (i >> 9);
        int kc = i & 511;
        Ag[b * NN + kc] = inbf[((size_t)(t + 1) * NB + b) * NIN + kc];
      }
    }

    grid_barrier(bar);       // panel fully updated before next TDM snapshot
  }
}

// -------------------------- host entry -------------------------------------

extern "C" void kernel_launch(void* const* d_in, const int* in_sizes, int n_in,
                              void* d_out, int out_size, void* d_ws, size_t ws_size,
                              hipStream_t stream) {
  (void)in_sizes; (void)n_in; (void)out_size; (void)ws_size;
  const float* in    = (const float*)d_in[0];   // (T,B,N) f32
  const float* W     = (const float*)d_in[1];   // (N,N)   f32
  const float* tau_m = (const float*)d_in[2];   // (N,)
  const float* tau_s = (const float*)d_in[3];   // (N,)
  const float* thr   = (const float*)d_in[4];   // (N,)
  float* out = (float*)d_out;

  char* ws = (char*)d_ws;
  unsigned*       bar  = (unsigned*)ws;                                 // 256 B
  unsigned short* Ag   = (unsigned short*)(ws + 256);                   // 128 KB
  unsigned short* inbf = (unsigned short*)(ws + 256 + 16 * NN * 2);     // 4 MB
  unsigned short* Wt   = (unsigned short*)(ws + 256 + 16 * NN * 2
                                              + (size_t)T_STEPS * NB * NIN * 2); // 32 MB

  const size_t nIn = (size_t)T_STEPS * NB * NIN;        // 2M elems
  k_init_input<<<(unsigned)((nIn + 255) / 256), 256, 0, stream>>>(in, inbf);

  dim3 tg(NN / 32, NN / 32), tb(32, 8);
  k_transpose<<<tg, tb, 0, stream>>>(W, Wt);

  k_init_A<<<(16 * NN + 255) / 256, 256, 0, stream>>>(Ag, inbf, bar);

  snn_persistent<<<NWG, 256, 0, stream>>>(inbf, Wt, Ag, tau_m, tau_s, thr, out, bar);
}
